// HTransformer1DAttention_69509750718578
// MI455X (gfx1250) — matrix operand order repaired and checked by reading, not compiled, not run on hardware
//
#include <hip/hip_runtime.h>
#include <hip/hip_bf16.h>

typedef __attribute__((ext_vector_type(16))) _Float16 v16h;
typedef __attribute__((ext_vector_type(8)))  float    v8f;

#define B_     4
#define N_     4096
#define H_     16
#define D_     64
#define HID_   1024
#define BH_    64
#define QKROWS_ 7936   // per-bh rows over qkv levels: 4096+2048+1024+512+256
#define YROWS_  12032  // per-bh rows over Y levels:  256+512+1024+2048+4096+4096

// ---------------------------------------------------------------------------
// CDNA5 async global->LDS copy (ASYNCcnt-tracked, bypasses VGPRs)
// ---------------------------------------------------------------------------
__device__ __forceinline__ unsigned lds_off_of(const void* p) {
  // Generic pointers into LDS carry the LDS byte offset in the low 32 bits.
  return (unsigned)(uintptr_t)p;
}

__device__ __forceinline__ void async_copy_b128(unsigned lds_byte_off, const void* gaddr) {
  asm volatile("global_load_async_to_lds_b128 %0, %1, off"
               :
               : "v"(lds_byte_off), "v"((unsigned long long)(uintptr_t)gaddr)
               : "memory");
}

__device__ __forceinline__ void wait_async0() {
  asm volatile("s_wait_asynccnt 0" ::: "memory");
}

// ---------------------------------------------------------------------------
// WMMA fragment helpers (contiguous k = 16*(lane>>4) + element for A and B, wave32)
// ---------------------------------------------------------------------------
__device__ __forceinline__ v8f wmma_f16(v16h a, v16h b, v8f c) {
  return __builtin_amdgcn_wmma_f32_16x16x32_f16(
      /*neg_a=*/false, a, /*neg_b=*/false, b,
      /*c_mod=*/(short)0, c, /*reuse_a=*/false, /*reuse_b=*/false);
}

// A fragment (16x32 f16): src row-major, ld elements.  lane&15 = M row.
__device__ __forceinline__ v16h load_frag_a(const _Float16* src, int ld, int lane) {
  v16h a;
  const int m  = lane & 15;
  const int hi = lane >> 4;
  const _Float16* row = src + (size_t)m * ld;
#pragma unroll
  for (int p = 0; p < 8; ++p) {
    int k = (hi << 4) + (p << 1); // hi*16+2*p
    a[2 * p]     = row[k];
    a[2 * p + 1] = row[k + 1];
  }
  return a;
}

// B fragment (32x16 f16), logical B[k][n] = src[k*ld + n]  (row-major K x N)
__device__ __forceinline__ v16h load_frag_b_rm(const _Float16* src, int ld, int lane) {
  v16h b;
  const int n  = lane & 15;
  const int hi = lane >> 4;
#pragma unroll
  for (int e = 0; e < 16; ++e)
    b[e] = src[(size_t)(hi * 16 + e) * ld + n];
  return b;
}

// B fragment (32x16 f16), logical B[k][n] = src[n*ld + k]  (transposed source)
__device__ __forceinline__ v16h load_frag_b_tn(const _Float16* src, int ld, int lane) {
  v16h b;
  const int n  = lane & 15;
  const int hi = lane >> 4;
  const _Float16* row = src + (size_t)n * ld + hi * 16;
#pragma unroll
  for (int e = 0; e < 16; ++e) b[e] = row[e];
  return b;
}

// ---------------------------------------------------------------------------
// LayerNorm: one 256-thread block per row of 1024, writes f16
// ---------------------------------------------------------------------------
__global__ __launch_bounds__(256) void ln_kernel(const float* __restrict__ hs,
                                                 const float* __restrict__ gamma,
                                                 const float* __restrict__ beta,
                                                 _Float16* __restrict__ x) {
  __shared__ float sh[256], sh2[256];
  const int row = blockIdx.x;
  const float* p = hs + (size_t)row * HID_;
  float s = 0.f, s2 = 0.f;
  for (int i = threadIdx.x; i < HID_; i += 256) {
    float v = p[i];
    s += v; s2 += v * v;
  }
  sh[threadIdx.x] = s; sh2[threadIdx.x] = s2;
  __syncthreads();
  for (int st = 128; st > 0; st >>= 1) {
    if (threadIdx.x < st) {
      sh[threadIdx.x]  += sh[threadIdx.x + st];
      sh2[threadIdx.x] += sh2[threadIdx.x + st];
    }
    __syncthreads();
  }
  const float mu  = sh[0] * (1.0f / HID_);
  const float var = sh2[0] * (1.0f / HID_) - mu * mu;
  const float rs  = rsqrtf(var + 1e-5f);
  for (int i = threadIdx.x; i < HID_; i += 256)
    x[(size_t)row * HID_ + i] = (_Float16)((p[i] - mu) * rs * gamma[i] + beta[i]);
}

// ---------------------------------------------------------------------------
// f32 -> f16 weight conversion
// ---------------------------------------------------------------------------
__global__ void cvt_kernel(const float* __restrict__ src, _Float16* __restrict__ dst, int n) {
  int i = blockIdx.x * blockDim.x + threadIdx.x;
  if (i < n) dst[i] = (_Float16)src[i];
}

// ---------------------------------------------------------------------------
// Generic f16 WMMA GEMM: C(MxN) = A(MxK) * B(KxN) [+ bias], block tile 128x64
// Tiles staged with CDNA5 async global->LDS copies.
// ---------------------------------------------------------------------------
#define TM 128
#define TN 64
#define TK 32

template <bool OUT_F16>
__global__ __launch_bounds__(128) void gemm_kernel(const _Float16* __restrict__ A,
                                                   const _Float16* __restrict__ Bm,
                                                   const float* __restrict__ bias,
                                                   void* __restrict__ Cout,
                                                   int M, int N, int K) {
  __shared__ __attribute__((aligned(16))) _Float16 As[TM][TK + 8]; // 80B rows
  __shared__ __attribute__((aligned(16))) _Float16 Bs[TK][TN + 8]; // 144B rows
  const int m0 = blockIdx.x * TM;
  const int n0 = blockIdx.y * TN;
  const int tid = threadIdx.x;
  const int wave = tid >> 5, lane = tid & 31;

  const unsigned asBase = lds_off_of(&As[0][0]);
  const unsigned bsBase = lds_off_of(&Bs[0][0]);

  v8f acc[2][4];
  v8f zero = {};
#pragma unroll
  for (int i = 0; i < 2; ++i)
#pragma unroll
    for (int j = 0; j < 4; ++j) acc[i][j] = zero;

  for (int k0 = 0; k0 < K; k0 += TK) {
    // A tile 128x32 halves = 512 b128 chunks -> 4 per thread (async to LDS)
#pragma unroll
    for (int i = 0; i < 4; ++i) {
      int idx = tid + i * 128;
      int r = idx >> 2;          // 4 chunks per row
      int c = (idx & 3) << 3;    // halves
      async_copy_b128(asBase + (unsigned)(r * (TK + 8) + c) * 2,
                      &A[(size_t)(m0 + r) * K + k0 + c]);
    }
    // B tile 32x64 halves = 256 b128 chunks -> 2 per thread
#pragma unroll
    for (int i = 0; i < 2; ++i) {
      int idx = tid + i * 128;
      int r = idx >> 3;          // 8 chunks per row
      int c = (idx & 7) << 3;
      async_copy_b128(bsBase + (unsigned)(r * (TN + 8) + c) * 2,
                      &Bm[(size_t)(k0 + r) * N + n0 + c]);
    }
    wait_async0();
    __syncthreads();

    v16h af[2], bf[4];
#pragma unroll
    for (int rt = 0; rt < 2; ++rt)
      af[rt] = load_frag_a(&As[wave * 32 + rt * 16][0], TK + 8, lane);
#pragma unroll
    for (int ct = 0; ct < 4; ++ct)
      bf[ct] = load_frag_b_rm(&Bs[0][ct * 16], TN + 8, lane);
#pragma unroll
    for (int rt = 0; rt < 2; ++rt)
#pragma unroll
      for (int ct = 0; ct < 4; ++ct)
        acc[rt][ct] = wmma_f16(af[rt], bf[ct], acc[rt][ct]);
    __syncthreads();
  }

  const int hi = lane >> 4, nn = lane & 15;
#pragma unroll
  for (int rt = 0; rt < 2; ++rt)
#pragma unroll
    for (int ct = 0; ct < 4; ++ct) {
      int rowb = m0 + wave * 32 + rt * 16 + hi * 8;
      int col  = n0 + ct * 16 + nn;
      float bv = bias ? bias[col] : 0.0f;
#pragma unroll
      for (int r = 0; r < 8; ++r) {
        float v = acc[rt][ct][r] + bv;
        if (OUT_F16)
          ((_Float16*)Cout)[(size_t)(rowb + r) * N + col] = (_Float16)v;
        else
          ((float*)Cout)[(size_t)(rowb + r) * N + col] = v;
      }
    }
}

// ---------------------------------------------------------------------------
// RoPE + scale + head split: qkv (B,N,3,H,D) f16 -> q,k,v (BH, N, D) f16
// ---------------------------------------------------------------------------
__global__ void rope_split_kernel(const _Float16* __restrict__ qkv,
                                  _Float16* __restrict__ q,
                                  _Float16* __restrict__ k,
                                  _Float16* __restrict__ v) {
  size_t idx = (size_t)blockIdx.x * blockDim.x + threadIdx.x; // B*N*H*32
  const int i = idx & 31;           // rotary pair index
  size_t t = idx >> 5;
  const int h = (int)(t & 15); t >>= 4;
  const int n = (int)(t & 4095);
  const int b = (int)(t >> 12);

  const size_t base = ((size_t)(b * N_ + n)) * (3 * HID_) + h * D_ + 2 * i;
  float q0 = (float)qkv[base],            q1 = (float)qkv[base + 1];
  float k0 = (float)qkv[base + HID_],     k1 = (float)qkv[base + HID_ + 1];
  float v0 = (float)qkv[base + 2 * HID_], v1 = (float)qkv[base + 2 * HID_ + 1];

  // inv = 1/(theta^(2i)/D) = D * theta^(-2i)   (faithful precedence)
  const float inv = (float)D_ * __powf(10000.0f, -2.0f * (float)i);
  const float fr = (float)n * inv;
  const float c = __cosf(fr), s = __sinf(fr);

  const float sc = 0.125f; // D^-0.5
  q0 *= sc; q1 *= sc;
  const float qo0 = q0 * c - q1 * s, qo1 = q1 * c + q0 * s;
  const float ko0 = k0 * c - k1 * s, ko1 = k1 * c + k0 * s;

  const size_t bh = (size_t)(b * H_ + h);
  const size_t o = (bh * QKROWS_ + n) * D_ + 2 * i; // finest level at offset 0
  q[o] = (_Float16)qo0; q[o + 1] = (_Float16)qo1;
  k[o] = (_Float16)ko0; k[o + 1] = (_Float16)ko1;
  v[o] = (_Float16)v0;  v[o + 1] = (_Float16)v1;
}

// ---------------------------------------------------------------------------
// Coarsen one level: q,k mean pairs; v sum pairs
// ---------------------------------------------------------------------------
__global__ void coarsen_kernel(_Float16* __restrict__ q, _Float16* __restrict__ k,
                               _Float16* __restrict__ v, int srcOff, int dstOff, int nOut) {
  size_t idx = (size_t)blockIdx.x * blockDim.x + threadIdx.x; // BH * nOut * D
  const int d = idx & 63;
  size_t t = idx >> 6;
  const int r  = (int)(t % nOut);
  const int bh = (int)(t / nOut);
  const size_t si = ((size_t)bh * QKROWS_ + srcOff + 2 * r) * D_ + d;
  const size_t di = ((size_t)bh * QKROWS_ + dstOff + r) * D_ + d;
  q[di] = (_Float16)(0.5f * ((float)q[si] + (float)q[si + D_]));
  k[di] = (_Float16)(0.5f * ((float)k[si] + (float)k[si + D_]));
  v[di] = (_Float16)((float)v[si] + (float)v[si + D_]);
}

// ---------------------------------------------------------------------------
// Block attention for one hierarchy level.
// One workgroup (4 waves) per (bh, qblock). flip: use neighbor block's k/v.
// q/k/v tiles staged via async global->LDS; exp(S) overlays q+k in LDS.
// ---------------------------------------------------------------------------
__global__ __launch_bounds__(128) void hattn_kernel(const _Float16* __restrict__ qb,
                                                    const _Float16* __restrict__ kb,
                                                    const _Float16* __restrict__ vb,
                                                    float* __restrict__ Y,
                                                    float* __restrict__ Asum,
                                                    int qkvOff, int yOff, int nblk, int flip) {
  __shared__ __attribute__((aligned(16))) _Float16 smem[3 * 128 * 72]; // 144B rows
  _Float16* qs = smem;
  _Float16* ks = smem + 128 * 72;
  _Float16* vs = smem + 2 * 128 * 72;
  _Float16* ss = smem; // 128 x 136 (272B rows), fits in q+k regions (17408 <= 18432)

  const int bq = blockIdx.x % nblk;
  const int bh = blockIdx.x / nblk;
  const int bk = flip ? (bq ^ 1) : bq;

  const _Float16* qg = qb + ((size_t)bh * QKROWS_ + qkvOff + bq * 128) * D_;
  const _Float16* kg = kb + ((size_t)bh * QKROWS_ + qkvOff + bk * 128) * D_;
  const _Float16* vg = vb + ((size_t)bh * QKROWS_ + qkvOff + bk * 128) * D_;

  const int tid = threadIdx.x, wave = tid >> 5, lane = tid & 31;
  const unsigned qsBase = lds_off_of(qs);
  const unsigned ksBase = lds_off_of(ks);
  const unsigned vsBase = lds_off_of(vs);

  // Stage 128x64 q,k,v tiles: 1024 b128 chunks each -> 8 per thread, async
#pragma unroll
  for (int i = 0; i < 8; ++i) {
    int idx = tid + i * 128;
    int r = idx >> 3;          // 8 chunks per row
    int c = (idx & 7) << 3;    // halves
    unsigned loff = (unsigned)(r * 72 + c) * 2;
    const size_t goff = (size_t)r * D_ + c;
    async_copy_b128(qsBase + loff, qg + goff);
    async_copy_b128(ksBase + loff, kg + goff);
    async_copy_b128(vsBase + loff, vg + goff);
  }
  wait_async0();
  __syncthreads();

  // S = q * k^T : each wave owns a 32-row strip, 2x8 tiles of 16x16
  v8f sacc[2][8];
  v8f zero = {};
#pragma unroll
  for (int rt = 0; rt < 2; ++rt)
#pragma unroll
    for (int ct = 0; ct < 8; ++ct) sacc[rt][ct] = zero;

#pragma unroll
  for (int kc = 0; kc < 64; kc += 32) {
    v16h af[2];
#pragma unroll
    for (int rt = 0; rt < 2; ++rt)
      af[rt] = load_frag_a(qs + (size_t)(wave * 32 + rt * 16) * 72 + kc, 72, lane);
#pragma unroll
    for (int ct = 0; ct < 8; ++ct) {
      v16h bf = load_frag_b_tn(ks + (size_t)(ct * 16) * 72 + kc, 72, lane);
#pragma unroll
      for (int rt = 0; rt < 2; ++rt)
        sacc[rt][ct] = wmma_f16(af[rt], bf, sacc[rt][ct]);
    }
  }

  // Row max -> exp -> row sum (16-lane shuffles stay within each half-wave)
  const int hi = lane >> 4, nn = lane & 15;
#pragma unroll
  for (int rt = 0; rt < 2; ++rt) {
#pragma unroll
    for (int r = 0; r < 8; ++r) {
      float mx = -3.0e38f;
#pragma unroll
      for (int ct = 0; ct < 8; ++ct) mx = fmaxf(mx, sacc[rt][ct][r]);
      mx = fmaxf(mx, __shfl_xor(mx, 1, 32));
      mx = fmaxf(mx, __shfl_xor(mx, 2, 32));
      mx = fmaxf(mx, __shfl_xor(mx, 4, 32));
      mx = fmaxf(mx, __shfl_xor(mx, 8, 32));
      float sum = 0.f;
#pragma unroll
      for (int ct = 0; ct < 8; ++ct) {
        float e = __expf(sacc[rt][ct][r] - mx);
        sacc[rt][ct][r] = e;
        sum += e;
      }
      sum += __shfl_xor(sum, 1, 32);
      sum += __shfl_xor(sum, 2, 32);
      sum += __shfl_xor(sum, 4, 32);
      sum += __shfl_xor(sum, 8, 32);
      if (nn == 0) {
        int row = wave * 32 + rt * 16 + hi * 8 + r;
        Asum[(size_t)bh * YROWS_ + yOff + bq * 128 + row] = sum;
      }
    }
  }

  __syncthreads(); // all waves done reading qs/ks before ss overlays them

  // exp(S) -> LDS as f16 (128 x 136)
#pragma unroll
  for (int rt = 0; rt < 2; ++rt)
#pragma unroll
    for (int ct = 0; ct < 8; ++ct) {
      int row = wave * 32 + rt * 16 + hi * 8;
#pragma unroll
      for (int r = 0; r < 8; ++r)
        ss[(size_t)(row + r) * 136 + ct * 16 + nn] = (_Float16)sacc[rt][ct][r];
    }
  __syncthreads();

  // Y = exp(S) * v : 2x4 tiles, K=128
  v8f yacc[2][4];
#pragma unroll
  for (int rt = 0; rt < 2; ++rt)
#pragma unroll
    for (int ct = 0; ct < 4; ++ct) yacc[rt][ct] = zero;

#pragma unroll
  for (int kc = 0; kc < 128; kc += 32) {
    v16h af[2];
#pragma unroll
    for (int rt = 0; rt < 2; ++rt)
      af[rt] = load_frag_a(ss + (size_t)(wave * 32 + rt * 16) * 136 + kc, 136, lane);
#pragma unroll
    for (int ct = 0; ct < 4; ++ct) {
      v16h bf = load_frag_b_rm(vs + (size_t)kc * 72 + ct * 16, 72, lane);
#pragma unroll
      for (int rt = 0; rt < 2; ++rt)
        yacc[rt][ct] = wmma_f16(af[rt], bf, yacc[rt][ct]);
    }
  }

  // Store Y (f32)
#pragma unroll
  for (int rt = 0; rt < 2; ++rt)
#pragma unroll
    for (int ct = 0; ct < 4; ++ct) {
      int rowb = wave * 32 + rt * 16 + hi * 8;
#pragma unroll
      for (int r = 0; r < 8; ++r)
        Y[((size_t)bh * YROWS_ + yOff + bq * 128 + rowb + r) * D_ + ct * 16 + nn] =
            yacc[rt][ct][r];
    }
}

// ---------------------------------------------------------------------------
// Combine the multi-resolution outputs: out = sum_l Y_l / (sum_l A_l + eps)
// ---------------------------------------------------------------------------
__global__ void combine_kernel(const float* __restrict__ Y, const float* __restrict__ Asum,
                               _Float16* __restrict__ attn) {
  size_t idx = (size_t)blockIdx.x * blockDim.x + threadIdx.x; // BH*N*D
  const int d = idx & 63;
  size_t t = idx >> 6;
  const int r  = (int)(t & 4095);
  const int bh = (int)(t >> 12);

  const int yoff[6]  = {0, 256, 768, 1792, 3840, 7936};
  const int shift[6] = {4, 3, 2, 1, 0, 0};
  const size_t base = (size_t)bh * YROWS_;
  float ys = 0.f, as = 0.f;
#pragma unroll
  for (int l = 0; l < 6; ++l) {
    int rr = yoff[l] + (r >> shift[l]);
    ys += Y[(base + rr) * D_ + d];
    as += Asum[base + rr];
  }
  const float o = ys / (as + 1e-8f);
  const int b = bh >> 4, h = bh & 15;
  attn[((size_t)(b * N_ + r)) * HID_ + h * D_ + d] = (_Float16)o;
}

// ---------------------------------------------------------------------------
// Host-side orchestration
// ---------------------------------------------------------------------------
extern "C" void kernel_launch(void* const* d_in, const int* in_sizes, int n_in,
                              void* d_out, int out_size, void* d_ws, size_t ws_size,
                              hipStream_t stream) {
  const float* hs    = (const float*)d_in[0];
  const float* gamma = (const float*)d_in[1];
  const float* beta  = (const float*)d_in[2];
  const float* wqkvf = (const float*)d_in[3];
  const float* woutf = (const float*)d_in[4];
  const float* boutf = (const float*)d_in[5];
  float* out = (float*)d_out;

  char* w = (char*)d_ws;
  auto carve = [&](size_t bytes) {
    char* p = w;
    w += (bytes + 255) & ~(size_t)255;
    return p;
  };
  const size_t MROWS = (size_t)B_ * N_; // 16384
  _Float16* x    = (_Float16*)carve(MROWS * HID_ * 2);
  _Float16* wqkv = (_Float16*)carve((size_t)HID_ * 3 * HID_ * 2);
  _Float16* wout = (_Float16*)carve((size_t)HID_ * HID_ * 2);
  _Float16* qkv  = (_Float16*)carve(MROWS * 3 * HID_ * 2);
  _Float16* q    = (_Float16*)carve((size_t)BH_ * QKROWS_ * D_ * 2);
  _Float16* k    = (_Float16*)carve((size_t)BH_ * QKROWS_ * D_ * 2);
  _Float16* v    = (_Float16*)carve((size_t)BH_ * QKROWS_ * D_ * 2);
  float*    Y    = (float*)   carve((size_t)BH_ * YROWS_ * D_ * 4);
  float*    A    = (float*)   carve((size_t)BH_ * YROWS_ * 4);
  _Float16* attn = (_Float16*)carve(MROWS * HID_ * 2);

  // 1. LayerNorm
  ln_kernel<<<MROWS, 256, 0, stream>>>(hs, gamma, beta, x);

  // 2. Weight conversion to f16
  {
    int nq = HID_ * 3 * HID_;
    cvt_kernel<<<(nq + 255) / 256, 256, 0, stream>>>(wqkvf, wqkv, nq);
    int no = HID_ * HID_;
    cvt_kernel<<<(no + 255) / 256, 256, 0, stream>>>(woutf, wout, no);
  }

  // 3. QKV GEMM (16384 x 1024) * (1024 x 3072)
  gemm_kernel<true><<<dim3(MROWS / TM, (3 * HID_) / TN), 128, 0, stream>>>(
      x, wqkv, nullptr, qkv, (int)MROWS, 3 * HID_, HID_);

  // 4. RoPE + scale + head split
  {
    size_t total = (size_t)B_ * N_ * H_ * 32;
    rope_split_kernel<<<total / 256, 256, 0, stream>>>(qkv, q, k, v);
  }

  // 5. Hierarchical coarsening: (src,dst,nOut)
  {
    const int srcOff[4] = {0, 4096, 6144, 7168};
    const int dstOff[4] = {4096, 6144, 7168, 7680};
    const int nOut[4]   = {2048, 1024, 512, 256};
    for (int l = 0; l < 4; ++l) {
      size_t total = (size_t)BH_ * nOut[l] * D_;
      coarsen_kernel<<<total / 256, 256, 0, stream>>>(q, k, v, srcOff[l], dstOff[l], nOut[l]);
    }
  }

  // 6. Block attention per level (coarsest -> finest flipped -> finest diag)
  {
    const int qkvOff[6] = {7680, 7168, 6144, 4096, 0, 0};
    const int yOff[6]   = {0, 256, 768, 1792, 3840, 7936};
    const int nblk[6]   = {2, 4, 8, 16, 32, 32};
    const int flip[6]   = {1, 1, 1, 1, 1, 0};
    for (int l = 0; l < 6; ++l) {
      hattn_kernel<<<BH_ * nblk[l], 128, 0, stream>>>(q, k, v, Y, A, qkvOff[l], yOff[l],
                                                      nblk[l], flip[l]);
    }
  }

  // 7. Combine levels, normalize, repack to (B, N, H*D)
  {
    size_t total = (size_t)BH_ * N_ * D_;
    combine_kernel<<<total / 256, 256, 0, stream>>>(Y, A, attn);
  }

  // 8. Output GEMM (16384 x 1024) * (1024 x 1024) + bias -> f32 d_out
  gemm_kernel<false><<<dim3(MROWS / TM, HID_ / TN), 128, 0, stream>>>(
      attn, wout, boutf, out, (int)MROWS, HID_, HID_);
}
